// Codebook_73581379715435
// MI455X (gfx1250) — matrix operand order repaired and checked by reading, not compiled, not run on hardware
//
#include <hip/hip_runtime.h>
#include <stdint.h>

// ---- CDNA5 wave32 WMMA types ----
typedef __attribute__((ext_vector_type(16))) __bf16 bf16x16;
typedef __attribute__((ext_vector_type(8)))  float  f32x8;
typedef __attribute__((ext_vector_type(4)))  float  f32x4;

union Frag {
  bf16x16 v;
  uint4   q[2];
};

__device__ __forceinline__ unsigned short f32_to_bf16(float f) {
  union { float f; unsigned u; } x; x.f = f;
  unsigned u = x.u;
  unsigned r = u + 0x7FFFu + ((u >> 16) & 1u);   // round-to-nearest-even
  return (unsigned short)(r >> 16);
}

// Problem constants
#define NB 32
#define NC 256
#define NH 32
#define NW 32
#define NM 2048
#define HW (NH * NW)

#define PIX 16          // pixels per workgroup
#define AS_STRIDE 264   // bf16 elements per A row in LDS (256 + 8 pad)
#define SD_STRIDE 2052  // f32 elements per dist row in LDS (2048 + 4 pad)

// -------- prep: templat f32 -> bf16, and t2[m] = ||templat_m||^2 --------
__global__ void codebook_prep(const float* __restrict__ templat,
                              float* __restrict__ t2,
                              unsigned short* __restrict__ tb) {
  const int m = blockIdx.x;
  const int t = threadIdx.x;                 // 256 threads = one row
  float v = templat[(size_t)m * NC + t];
  tb[(size_t)m * NC + t] = f32_to_bf16(v);
  float s = v * v;
  #pragma unroll
  for (int off = 16; off > 0; off >>= 1) s += __shfl_xor(s, off);
  __shared__ float part[8];
  if ((t & 31) == 0) part[t >> 5] = s;
  __syncthreads();
  if (t == 0) {
    float tot = 0.f;
    #pragma unroll
    for (int i = 0; i < 8; ++i) tot += part[i];
    t2[m] = tot;
  }
}

// -------- main: GEMM (bf16 WMMA) + fused softmax + argmax --------
__global__ __launch_bounds__(256)
void codebook_main(const float* __restrict__ encode,
                   const float* __restrict__ t2,
                   const unsigned short* __restrict__ tb,
                   float* __restrict__ zsoft,
                   int* __restrict__ zidx) {
  extern __shared__ char smem[];
  unsigned short* As  = (unsigned short*)smem;                              // 16 x 264 bf16
  float*          Sd  = (float*)(smem + PIX * AS_STRIDE * 2);               // 16 x 2052 f32
  float*          Ssum= (float*)(smem + PIX * AS_STRIDE * 2 + PIX * SD_STRIDE * 4);

  const int tid  = threadIdx.x;
  const int wave = tid >> 5;
  const int lane = tid & 31;
  const int half = lane >> 4;
  const int row  = lane & 15;

  const int blk = blockIdx.x;                 // 2048 blocks
  const int b   = blk >> 6;                   // 64 blocks per batch image
  const int r6  = blk & 63;
  const int h   = r6 >> 1;
  const int w0  = (r6 & 1) << 4;              // 16 consecutive w pixels
  const int hwbase = h * NW + w0;

  // ---- stage A tile: 16 pixels x 256 channels, f32 (strided) -> bf16 LDS ----
  // encode is read exactly once grid-wide -> non-temporal loads keep L2 for templat
  {
    const float* ebase = encode + ((size_t)b * NC) * HW + hwbase;
    const int px = tid & 15;
    const int c0 = tid >> 4;
    #pragma unroll
    for (int i = 0; i < 16; ++i) {
      int c = c0 + i * 16;                                   // coalesced: px fastest
      float v = __builtin_nontemporal_load(&ebase[(size_t)c * HW + px]);
      As[px * AS_STRIDE + c] = f32_to_bf16(v);
    }
  }
  __syncthreads();

  // ---- A fragments: ISA 16-bit A layout (lane<16: K 0-7 & 16-23; lane>=16: +8) ----
  Frag af[8];
  #pragma unroll
  for (int kc = 0; kc < 8; ++kc) {
    const unsigned short* ap = &As[row * AS_STRIDE + kc * 32];
    af[kc].q[0] = *(const uint4*)(ap + half * 8);
    af[kc].q[1] = *(const uint4*)(ap + 16 + half * 8);
  }

  // ---- GEMM: each wave covers m-tiles {wave + 8j | j=0..15}, K=256 in 8 steps ----
  for (int j = 0; j < 16; ++j) {
    const int mb = (wave + 8 * j) * 16;
    // B layout (dense 32x16): lanes 0-15 hold K 0-7 & 16-23, lanes 16-31 hold K 8-15 & 24-31,
    // lane = output column n = templat row -> two 16B loads per K-chunk.
    const unsigned short* brow = tb + (size_t)(mb + row) * NC + half * 8;
    f32x8 acc = {0.f, 0.f, 0.f, 0.f, 0.f, 0.f, 0.f, 0.f};
    #pragma unroll
    for (int kc = 0; kc < 8; ++kc) {
      Frag bf;
      bf.q[0] = *(const uint4*)(brow + kc * 32);
      bf.q[1] = *(const uint4*)(brow + kc * 32 + 16);
      acc = __builtin_amdgcn_wmma_f32_16x16x32_bf16(
          false, af[kc].v, false, bf.v, (short)0, acc, false, false);
    }
    // logit s = 2*cross - t2 (e2 cancels in softmax/argmax)
    const float t2v = t2[mb + row];
    #pragma unroll
    for (int r = 0; r < 8; ++r) {
      // C/D layout: VGPR r, lanes 0-15 -> M=r, lanes 16-31 -> M=r+8; N = lane&15
      Sd[(r + half * 8) * SD_STRIDE + mb + row] = 2.0f * acc[r] - t2v;
    }
  }
  __syncthreads();

  // ---- softmax reductions: 2 pixels per wave ----
  #pragma unroll
  for (int pp = 0; pp < 2; ++pp) {
    const int p = wave * 2 + pp;
    float* srow = &Sd[p * SD_STRIDE];
    float bmax = -3.402823466e38f;
    int   bidx = 0;
    for (int i = lane; i < NM; i += 32) {
      float v = srow[i];
      if (v > bmax) { bmax = v; bidx = i; }    // strict > keeps first occurrence
    }
    #pragma unroll
    for (int off = 16; off > 0; off >>= 1) {
      float ov = __shfl_xor(bmax, off);
      int   oi = __shfl_xor(bidx, off);
      if (ov > bmax || (ov == bmax && oi < bidx)) { bmax = ov; bidx = oi; }
    }
    float psum = 0.f;
    for (int i = lane; i < NM; i += 32) {
      float e = __expf(srow[i] - bmax);
      srow[i] = e;
      psum += e;
    }
    #pragma unroll
    for (int off = 16; off > 0; off >>= 1) psum += __shfl_xor(psum, off);
    if (lane == 0) {
      Ssum[p] = psum;
      zidx[(size_t)b * HW + hwbase + p] = bidx;
    }
  }
  __syncthreads();

  // ---- normalized write: float4 over 4 consecutive pixels, non-temporal ----
  // lanes 0-3 cover all 16 px for one m -> 64B contiguous per m; NT keeps L2 clean.
  {
    const int pg = tid & 3;                   // pixel group: px = pg*4 .. pg*4+3
    const int m0 = tid >> 2;                  // 64 m-slices
    const int px0 = pg * 4;
    f32x4 inv;
    #pragma unroll
    for (int i = 0; i < 4; ++i) inv[i] = 1.0f / Ssum[px0 + i];
    float* obase = zsoft + ((size_t)b * NM) * HW + hwbase + px0;
    const float* srow = &Sd[px0 * SD_STRIDE];
    for (int m = m0; m < NM; m += 64) {
      f32x4 v;
      #pragma unroll
      for (int i = 0; i < 4; ++i) v[i] = srow[i * SD_STRIDE + m] * inv[i];
      __builtin_nontemporal_store(v, (f32x4*)(obase + (size_t)m * HW));
    }
  }
}

extern "C" void kernel_launch(void* const* d_in, const int* in_sizes, int n_in,
                              void* d_out, int out_size, void* d_ws, size_t ws_size,
                              hipStream_t stream) {
  const float* encode  = (const float*)d_in[0];   // (32,256,32,32) f32
  const float* templat = (const float*)d_in[1];   // (2048,256) f32

  float*          t2 = (float*)d_ws;                                  // 2048 f32
  unsigned short* tb = (unsigned short*)((char*)d_ws + 8192);         // 2048x256 bf16 (1 MB)

  float* zsoft = (float*)d_out;                                       // (32,2048,32,32)
  int*   zidx  = (int*)(zsoft + (size_t)NB * NM * HW);                // (32,32,32) i32

  codebook_prep<<<NM, 256, 0, stream>>>(templat, t2, tb);

  const size_t smem = (size_t)PIX * AS_STRIDE * 2   // A tile bf16
                    + (size_t)PIX * SD_STRIDE * 4   // dist/exp rows f32
                    + (size_t)PIX * 4;              // per-pixel sums
  codebook_main<<<(NB * NH * NW) / PIX, 256, smem, stream>>>(encode, t2, tb, zsoft, zidx);
}
